// SharedCoreExperts_17600775979483
// MI455X (gfx1250) — matrix operand order repaired and checked by reading, not compiled, not run on hardware
//
#include <hip/hip_runtime.h>
#include <hip/hip_bf16.h>
#include <stddef.h>

// ---------------------------------------------------------------------------
// LoRA-MoE on gfx1250: sort tokens by expert, pre-convert to bf16 once,
// then dense per-expert v_wmma_f32_16x16x32_bf16 chain. Interior X tiles
// stream via global_load_async_to_lds_b128 (double-buffered, ASYNCcnt);
// all fragment movement is b128; epilogues staged through LDS.
// ---------------------------------------------------------------------------

#define T_TOK 4096
#define DH    2048
#define DI    5632
#define RNK   128
#define NEXP  8

typedef __attribute__((ext_vector_type(16))) __bf16 v16bf;
typedef __attribute__((ext_vector_type(8)))  float  v8f;
typedef __attribute__((ext_vector_type(2)))  float  v2f;
typedef __attribute__((ext_vector_type(2)))  __bf16 v2bf;

union BF16x16 {
  v16bf v;
  unsigned int u[8];
};

// fp32 pair -> packed bf16 pair. Prefer the dedicated builtin; otherwise use a
// vector fptrunc, which the gfx1250 backend selects to v_cvt_pk_bf16_f32.
#if __has_builtin(__builtin_amdgcn_cvt_pk_bf16_f32)
__device__ inline unsigned int pack2(float lo, float hi) {
  auto r = __builtin_amdgcn_cvt_pk_bf16_f32(lo, hi);
  union { decltype(r) v; unsigned int u; } cv;
  cv.v = r;
  return cv.u;
}
#else
__device__ inline unsigned int pack2(float lo, float hi) {
  v2f in;
  in.x = lo;
  in.y = hi;
  v2bf r = __builtin_convertvector(in, v2bf);
  union { v2bf v; unsigned int u; } cv;
  cv.v = r;
  return cv.u;
}
#endif

__device__ inline unsigned short bf16bits(float f) {
  union { __bf16 b; unsigned short s; } cv;
  cv.b = (__bf16)f;
  return cv.s;
}

__device__ inline unsigned lds_off(const void* p) {
  return (unsigned)(unsigned long long)p;
}

__device__ inline void load_afrag(BF16x16& f, const unsigned short* base) {
  uint4 a = *(const uint4*)(base);
  uint4 b = *(const uint4*)(base + 16);
  f.u[0] = a.x; f.u[1] = a.y; f.u[2] = a.z; f.u[3] = a.w;
  f.u[4] = b.x; f.u[5] = b.y; f.u[6] = b.z; f.u[7] = b.w;
}
__device__ inline void load_bfrag(BF16x16& f, const unsigned short* p) {
  uint4 a = *(const uint4*)(p);
  uint4 b = *(const uint4*)(p + 16);
  f.u[0] = a.x; f.u[1] = a.y; f.u[2] = a.z; f.u[3] = a.w;
  f.u[4] = b.x; f.u[5] = b.y; f.u[6] = b.z; f.u[7] = b.w;
}

#define WMMA_BF16(A, B, C) \
  __builtin_amdgcn_wmma_f32_16x16x32_bf16(false, (A).v, false, (B).v, (short)0, (C), false, false)

// ---------------------------------------------------------------------------
__global__ void moe_init(int* counts) {
  if (threadIdx.x < NEXP) counts[threadIdx.x] = 0;
}

__global__ void moe_count(const int* __restrict__ idx, int* __restrict__ counts) {
  int t = blockIdx.x * 256 + threadIdx.x;
  if (t < T_TOK) atomicAdd(&counts[idx[t]], 1);
}

__global__ void moe_scan(const int* __restrict__ counts, int* __restrict__ off) {
  if (threadIdx.x == 0) {
    int a = 0;
    for (int e = 0; e < NEXP; ++e) { off[e] = a; a += counts[e]; }
    off[NEXP] = a;
  }
}

__global__ void moe_scatter(const int* __restrict__ idx, const int* __restrict__ off,
                            int* __restrict__ perm) {
  int wave = threadIdx.x >> 5;
  int lane = threadIdx.x & 31;
  int base = off[wave];
  for (int t0 = 0; t0 < T_TOK; t0 += 32) {
    int t = t0 + lane;
    bool pred = (idx[t] == wave);
    unsigned int mask = (unsigned int)__ballot(pred);
    unsigned int prefix = __popc(mask & ((1u << lane) - 1u));
    if (pred) perm[base + prefix] = t;
    base += __popc(mask);
  }
}

// ---------------------------------------------------------------------------
__global__ __launch_bounds__(256) void cvt_f32_bf16(
    const float* __restrict__ src, unsigned short* __restrict__ dst, long n) {
  long i = ((long)blockIdx.x * 256 + threadIdx.x) * 4;
  if (i >= n) return;
  float4 v = *(const float4*)(src + i);
  uint2 r;
  r.x = pack2(v.x, v.y);
  r.y = pack2(v.z, v.w);
  *(uint2*)(dst + i) = r;
}

__global__ __launch_bounds__(256) void gather_x_bf16(
    const float* __restrict__ X, const int* __restrict__ perm,
    unsigned short* __restrict__ Xg) {
  long i = ((long)blockIdx.x * 256 + threadIdx.x) * 4;
  if (i >= (long)T_TOK * DH) return;
  int p = (int)(i / DH);
  int c = (int)(i % DH);
  int t = perm[p];
  float4 v = *(const float4*)(X + (size_t)t * DH + c);
  uint2 r;
  r.x = pack2(v.x, v.y);
  r.y = pack2(v.z, v.w);
  *(uint2*)(Xg + i) = r;
}

// ---------------------------------------------------------------------------
// Stage A+C:  H2[p,0..127] = (X[p,:] @ A_e^T) @ C^T  for a 64-token tile.
// ---------------------------------------------------------------------------
__global__ __launch_bounds__(256) void stage_ac(
    const unsigned short* __restrict__ Xin,
    int Din,
    const unsigned short* __restrict__ Abf,
    const unsigned short* __restrict__ Cbf,
    const int* __restrict__ off, unsigned short* __restrict__ Hout)
{
  __shared__ unsigned short Xs[2 * 64 * 40];
  __shared__ unsigned short H1s[64 * 136];

  const int e     = blockIdx.y;
  const int seg_e = off[e + 1];
  const int m0    = off[e] + blockIdx.x * 64;
  if (m0 >= seg_e) return;
  const bool full = (m0 + 64 <= seg_e);

  const int tid  = threadIdx.x;
  const int wave = tid >> 5, lane = tid & 31;
  const int hf   = lane >> 4, nl = lane & 15;
  const int rbase = wave * 16;

  const int lm = tid >> 2, lc = tid & 3;
  const int lrow = m0 + lm;
  const unsigned short* gptr = Xin + (size_t)lrow * Din + lc * 8;
  const unsigned lbase = lds_off(&Xs[lm * 40 + lc * 8]);

  v8f acc[4] = {};
  const unsigned short* Arow = Abf + ((size_t)e * RNK + rbase + nl) * (size_t)Din;
  const int nIter = Din / 32;

  auto compute = [&](int it, int cur) {
    BF16x16 bw;
    load_bfrag(bw, Arow + it * 32 + 8 * hf);
    #pragma unroll
    for (int s = 0; s < 4; ++s) {
      BF16x16 af;
      load_afrag(af, &Xs[cur * 2560 + (16 * s + nl) * 40 + 8 * hf]);
      acc[s] = WMMA_BF16(af, bw, acc[s]);
    }
  };

  if (full) {
    // interior: async DMA to LDS, double-buffered, one barrier per iteration
    auto async_load = [&](int k0, int buf) {
      unsigned long long g = (unsigned long long)(gptr + k0);
      unsigned l = lbase + buf * 5120;
      asm volatile("global_load_async_to_lds_b128 %0, %1, off"
                   :: "v"(l), "v"(g) : "memory");
    };
    async_load(0, 0);
    asm volatile("s_wait_asynccnt 0" ::: "memory");
    __syncthreads();
    for (int it = 0; it < nIter; ++it) {
      const int cur = it & 1;
      if (it + 1 < nIter) async_load((it + 1) * 32, cur ^ 1);
      compute(it, cur);
      asm volatile("s_wait_asynccnt 0" ::: "memory");
      __syncthreads();
    }
  } else {
    // segment edge: guarded manual path with zero fill
    auto manual_load = [&](int k0, int buf) {
      uint4 d = make_uint4(0u, 0u, 0u, 0u);
      if (lrow < seg_e) d = *(const uint4*)(gptr + k0);
      *(uint4*)&Xs[buf * 2560 + lm * 40 + lc * 8] = d;
    };
    manual_load(0, 0);
    __syncthreads();
    for (int it = 0; it < nIter; ++it) {
      const int cur = it & 1;
      if (it + 1 < nIter) manual_load((it + 1) * 32, cur ^ 1);
      compute(it, cur);
      __syncthreads();
    }
  }

  #pragma unroll
  for (int s = 0; s < 4; ++s)
    #pragma unroll
    for (int i = 0; i < 8; ++i)
      H1s[(16 * s + i + 8 * hf) * 136 + rbase + nl] = bf16bits(acc[s][i]);
  __syncthreads();

  v8f acc2[4] = {};
  const unsigned short* Crow = Cbf + (size_t)(rbase + nl) * RNK;
  #pragma unroll
  for (int kc = 0; kc < RNK; kc += 32) {
    BF16x16 bw;
    load_bfrag(bw, Crow + kc + 8 * hf);
    #pragma unroll
    for (int s = 0; s < 4; ++s) {
      BF16x16 af;
      load_afrag(af, &H1s[(16 * s + nl) * 136 + kc + 8 * hf]);
      acc2[s] = WMMA_BF16(af, bw, acc2[s]);
    }
  }
  __syncthreads();

  #pragma unroll
  for (int s = 0; s < 4; ++s)
    #pragma unroll
    for (int i = 0; i < 8; ++i)
      H1s[(16 * s + i + 8 * hf) * 136 + rbase + nl] = bf16bits(acc2[s][i]);
  __syncthreads();

  #pragma unroll
  for (int rep = 0; rep < 4; ++rep) {
    int g = tid + rep * 256;
    int m = g >> 4, c = g & 15;
    if (m0 + m < seg_e)
      *(uint4*)(Hout + (size_t)(m0 + m) * RNK + c * 8) =
          *(const uint4*)&H1s[m * 136 + c * 8];
  }
}

// ---------------------------------------------------------------------------
__global__ __launch_bounds__(256) void stage_b_gateup(
    const unsigned short* __restrict__ Hg, const unsigned short* __restrict__ Hu,
    const unsigned short* __restrict__ gateB, const unsigned short* __restrict__ upB,
    const int* __restrict__ off, unsigned short* __restrict__ Z)
{
  __shared__ unsigned short HgS[64 * 136];
  __shared__ unsigned short HuS[64 * 136];

  const int e     = blockIdx.z;
  const int seg_e = off[e + 1];
  const int m0    = off[e] + blockIdx.y * 64;
  if (m0 >= seg_e) return;

  const int tid  = threadIdx.x;
  const int wave = tid >> 5, lane = tid & 31;
  const int hf   = lane >> 4, nl = lane & 15;
  const int nbase = blockIdx.x * 128;
  const int ncol  = nbase + wave * 16 + nl;

  #pragma unroll
  for (int rep = 0; rep < 4; ++rep) {
    int g = tid + rep * 256;
    int m = g >> 4, c = g & 15;
    size_t gi = (size_t)(m0 + m) * RNK + c * 8;
    *(uint4*)&HgS[m * 136 + c * 8] = *(const uint4*)(Hg + gi);
    *(uint4*)&HuS[m * 136 + c * 8] = *(const uint4*)(Hu + gi);
  }
  __syncthreads();

  v8f accg[4] = {}, accu[4] = {};
  const unsigned short* gB = gateB + ((size_t)e * DI + ncol) * (size_t)RNK;
  const unsigned short* uB = upB   + ((size_t)e * DI + ncol) * (size_t)RNK;

  #pragma unroll
  for (int kc = 0; kc < RNK; kc += 32) {
    BF16x16 bwg, bwu;
    load_bfrag(bwg, gB + kc + 8 * hf);
    load_bfrag(bwu, uB + kc + 8 * hf);
    #pragma unroll
    for (int s = 0; s < 4; ++s) {
      BF16x16 ag, au;
      load_afrag(ag, &HgS[(16 * s + nl) * 136 + kc + 8 * hf]);
      load_afrag(au, &HuS[(16 * s + nl) * 136 + kc + 8 * hf]);
      accg[s] = WMMA_BF16(ag, bwg, accg[s]);
      accu[s] = WMMA_BF16(au, bwu, accu[s]);
    }
  }
  __syncthreads();

  #pragma unroll
  for (int s = 0; s < 4; ++s)
    #pragma unroll
    for (int i = 0; i < 8; ++i) {
      float g  = accg[s][i];
      float sl = g / (1.f + __expf(-g));
      HgS[(16 * s + i + 8 * hf) * 136 + wave * 16 + nl] = bf16bits(sl * accu[s][i]);
    }
  __syncthreads();

  #pragma unroll
  for (int rep = 0; rep < 4; ++rep) {
    int g = tid + rep * 256;
    int m = g >> 4, c = g & 15;
    if (m0 + m < seg_e)
      *(uint4*)(Z + (size_t)(m0 + m) * DI + nbase + c * 8) =
          *(const uint4*)&HgS[m * 136 + c * 8];
  }
}

// ---------------------------------------------------------------------------
__global__ __launch_bounds__(256) void stage_b_down(
    const unsigned short* __restrict__ Hd, const unsigned short* __restrict__ downB,
    const float* __restrict__ bias, const int* __restrict__ off,
    const int* __restrict__ perm, float* __restrict__ Out)
{
  __shared__ unsigned short HdS[64 * 136];
  __shared__ float OutS[64 * 132];
  __shared__ int PermS[64];

  const int e     = blockIdx.z;
  const int seg_e = off[e + 1];
  const int m0    = off[e] + blockIdx.y * 64;
  if (m0 >= seg_e) return;

  const int tid  = threadIdx.x;
  const int wave = tid >> 5, lane = tid & 31;
  const int hf   = lane >> 4, nl = lane & 15;
  const int nbase = blockIdx.x * 128;
  const int ncol  = nbase + wave * 16 + nl;

  if (tid < 64) PermS[tid] = (m0 + tid < seg_e) ? perm[m0 + tid] : 0;
  #pragma unroll
  for (int rep = 0; rep < 4; ++rep) {
    int g = tid + rep * 256;
    int m = g >> 4, c = g & 15;
    *(uint4*)&HdS[m * 136 + c * 8] = *(const uint4*)(Hd + (size_t)(m0 + m) * RNK + c * 8);
  }
  __syncthreads();

  v8f acc[4] = {};
  const unsigned short* dB = downB + ((size_t)e * DH + ncol) * (size_t)RNK;

  #pragma unroll
  for (int kc = 0; kc < RNK; kc += 32) {
    BF16x16 bw;
    load_bfrag(bw, dB + kc + 8 * hf);
    #pragma unroll
    for (int s = 0; s < 4; ++s) {
      BF16x16 af;
      load_afrag(af, &HdS[(16 * s + nl) * 136 + kc + 8 * hf]);
      acc[s] = WMMA_BF16(af, bw, acc[s]);
    }
  }

  float b = bias[(size_t)e * DH + ncol];
  #pragma unroll
  for (int s = 0; s < 4; ++s)
    #pragma unroll
    for (int i = 0; i < 8; ++i)
      OutS[(16 * s + i + 8 * hf) * 132 + wave * 16 + nl] = acc[s][i] + b;
  __syncthreads();

  #pragma unroll
  for (int rep = 0; rep < 8; ++rep) {
    int g = tid + rep * 256;
    int m = g >> 5, c = g & 31;
    if (m0 + m < seg_e)
      *(float4*)(Out + (size_t)PermS[m] * DH + nbase + c * 4) =
          *(const float4*)&OutS[m * 132 + c * 4];
  }
}

// ---------------------------------------------------------------------------
extern "C" void kernel_launch(void* const* d_in, const int* in_sizes, int n_in,
                              void* d_out, int out_size, void* d_ws, size_t ws_size,
                              hipStream_t stream) {
  (void)in_sizes; (void)n_in; (void)out_size; (void)ws_size;

  const float* x         = (const float*)d_in[0];
  const int*   eidx      = (const int*)  d_in[1];
  const float* gate_A    = (const float*)d_in[2];
  const float* gate_C    = (const float*)d_in[3];
  const float* gate_B    = (const float*)d_in[4];
  const float* up_A      = (const float*)d_in[5];
  const float* up_C      = (const float*)d_in[6];
  const float* up_B      = (const float*)d_in[7];
  const float* down_A    = (const float*)d_in[8];
  const float* down_C    = (const float*)d_in[9];
  const float* down_B    = (const float*)d_in[10];
  const float* down_bias = (const float*)d_in[11];
  float* out = (float*)d_out;

  char* ws = (char*)d_ws;
  int* counts = (int*)ws;
  int* off    = (int*)(ws + 64);
  int* perm   = (int*)(ws + 256);
  size_t o = 256 + (size_t)T_TOK * 4;
  o = (o + 255) & ~(size_t)255;
  auto take = [&](size_t nelem) {
    unsigned short* p = (unsigned short*)(ws + o);
    o += nelem * 2;
    o = (o + 255) & ~(size_t)255;
    return p;
  };
  const long nA  = (long)NEXP * RNK * DH;
  const long nAd = (long)NEXP * RNK * DI;
  const long nC  = (long)RNK * RNK;

  unsigned short* Xg  = take((size_t)T_TOK * DH);
  unsigned short* wAg = take(nA);
  unsigned short* wAu = take(nA);
  unsigned short* wAd = take(nAd);
  unsigned short* wBg = take(nAd);
  unsigned short* wBu = take(nAd);
  unsigned short* wBd = take(nA);
  unsigned short* wCg = take(nC);
  unsigned short* wCu = take(nC);
  unsigned short* wCd = take(nC);
  unsigned short* Hg  = take((size_t)T_TOK * RNK);
  unsigned short* Hu  = take((size_t)T_TOK * RNK);
  unsigned short* Hd  = take((size_t)T_TOK * RNK);
  unsigned short* Z   = take((size_t)T_TOK * DI);

  moe_init   <<<1, 32, 0, stream>>>(counts);
  moe_count  <<<(T_TOK + 255) / 256, 256, 0, stream>>>(eidx, counts);
  moe_scan   <<<1, 32, 0, stream>>>(counts, off);
  moe_scatter<<<1, 256, 0, stream>>>(eidx, off, perm);

  auto cvt = [&](const float* s, unsigned short* d, long n) {
    cvt_f32_bf16<<<(unsigned)((n / 4 + 255) / 256), 256, 0, stream>>>(s, d, n);
  };
  cvt(gate_A, wAg, nA);   cvt(up_A, wAu, nA);   cvt(down_A, wAd, nAd);
  cvt(gate_B, wBg, nAd);  cvt(up_B, wBu, nAd);  cvt(down_B, wBd, nA);
  cvt(gate_C, wCg, nC);   cvt(up_C, wCu, nC);   cvt(down_C, wCd, nC);
  gather_x_bf16<<<(unsigned)(((long)T_TOK * DH / 4 + 255) / 256), 256, 0, stream>>>(
      x, perm, Xg);

  dim3 gAC(T_TOK / 64, NEXP);
  stage_ac<<<gAC, 256, 0, stream>>>(Xg, DH, wAg, wCg, off, Hg);
  stage_ac<<<gAC, 256, 0, stream>>>(Xg, DH, wAu, wCu, off, Hu);
  stage_b_gateup<<<dim3(DI / 128, T_TOK / 64, NEXP), 256, 0, stream>>>(
      Hg, Hu, wBg, wBu, off, Z);
  stage_ac<<<gAC, 256, 0, stream>>>(Z, DI, wAd, wCd, off, Hd);
  stage_b_down<<<dim3(DH / 128, T_TOK / 64, NEXP), 256, 0, stream>>>(
      Hd, wBd, down_bias, off, perm, out);
}
